// AdversarialAttention_54503134986461
// MI455X (gfx1250) — matrix operand order repaired and checked by reading, not compiled, not run on hardware
//
#include <hip/hip_runtime.h>
#include <hip/hip_bf16.h>

// ---------------------------------------------------------------------------
// AdversarialAttention on MI455X (gfx1250), bf16 WMMA, flash-style fusion.
//   B=2, S=2048, E=1024, H=16, DH=64
// Pipeline:
//   1) gemm<0,false>: Q = query*Wq^T+bq  -> bf16 [B,H,S,DH]
//   2) gemm<0,false>: K = key  *Wk^T+bk  -> bf16 [B,H,S,DH]
//   3) gemm<1,false>: V = value*Wv^T+bv  -> bf16 [B,H,DH,S]  (transposed)
//   4) attn_fused: per (b, 32-query tile), 16 waves = 16 heads; scores kept
//      in LDS; cross-head logit mixing (I+Wc)+bc done with
//      V_WMMA_F32_16X16X4_F32 (M=16 heads, K=16 heads, N=positions);
//      online softmax; P*V via bf16 WMMA. V fragments fetched during the
//      softmax phase to hide HBM latency.
//   5) gemm<2,true>:  out = AO*Wo^T+bo   -> f32 d_out
// ---------------------------------------------------------------------------

typedef __bf16        v16bf __attribute__((ext_vector_type(16)));
typedef float         v8f   __attribute__((ext_vector_type(8)));
typedef float         v2f   __attribute__((ext_vector_type(2)));
typedef unsigned int  u32x4 __attribute__((ext_vector_type(4)));
typedef float         f32x4 __attribute__((ext_vector_type(4)));

union Frag { v16bf bf; u32x4 u[2]; };

__device__ __forceinline__ v8f vzero8() {
    v8f z;
#pragma unroll
    for (int i = 0; i < 8; ++i) z[i] = 0.0f;
    return z;
}

__device__ __forceinline__ v8f wmma_bf16(const Frag& a, const Frag& b, v8f c) {
    // (neg_a, A, neg_b, B, c_mod, C, reuse_a, reuse_b)
    return __builtin_amdgcn_wmma_f32_16x16x32_bf16(false, a.bf, false, b.bf,
                                                   (short)0, c, false, false);
}

__device__ __forceinline__ v8f wmma_f32k4(v2f a, v2f b, v8f c) {
    // V_WMMA_F32_16X16X4_F32: A 16x4, B 4x16, 2 VGPRs each.
    return __builtin_amdgcn_wmma_f32_16x16x4_f32(false, a, false, b,
                                                 (short)0, c, false, false);
}

constexpr int kB  = 2;
constexpr int kS  = 2048;
constexpr int kE  = 1024;
constexpr int kH  = 16;
constexpr int kDH = 64;

// ---------------------------------------------------------------------------
// GEMM: C[M,N] = A[M,K] * W[N,K]^T + bias[N]
//  block = 256 thr (8 waves), tile 128(M) x 128(N), k-step 32, bf16 WMMA,
//  double-buffered LDS (2 x (128x32 A + 128x32 W) bf16 = 32 KB).
//  Wave tile 32x64: 8 WMMAs / 6 fragment loads per k-step.
//  OUT_MODE 0: bf16 -> [B,H,S,DH]   (Q/K head-major)
//  OUT_MODE 1: bf16 -> [B,H,DH,S]   (V transposed for P*V fragment loads)
//  OUT_MODE 2: f32  -> [M,E]        (final output)
// ---------------------------------------------------------------------------
template <int OUT_MODE, bool A_BF16>
__global__ __launch_bounds__(256) void gemm_bf16_wmma(
    const void* __restrict__ Ap, const float* __restrict__ W,
    const float* __restrict__ bias, void* __restrict__ Outp)
{
    extern __shared__ char smem[];
    __bf16* AsBase = (__bf16*)smem;              // [2][128][32] bf16
    __bf16* WsBase = (__bf16*)(smem + 16384);    // [2][128][32] bf16

    const int tid  = threadIdx.x;
    const int wid  = tid >> 5;
    const int lane = tid & 31;
    const int ln   = lane & 15;
    const int hi   = lane >> 4;

    const int nblk = blockIdx.x & 7;      // 8 n-blocks of 128
    const int mblk = blockIdx.x >> 3;     // 32 m-blocks of 128
    const int m0 = mblk * 128, n0 = nblk * 128;
    const int mw0 = (wid & 3) * 32;       // wave sub-tile 32(M) x 64(N)
    const int nw0 = (wid >> 2) * 64;

    // staging registers for the in-flight k-tile
    f32x4 aF[4]; u32x4 aU[2]; f32x4 wF[4];

    auto load_tile = [&](int kb) {
        const int k0 = kb * 32;
        if constexpr (A_BF16) {
            const __bf16* A = (const __bf16*)Ap;
#pragma unroll
            for (int j = 0; j < 2; ++j) {
                int i = tid + 256 * j;              // 128 rows x 4 x 16B
                int row = i >> 2, seg = i & 3;
                aU[j] = *(const u32x4*)(A + (size_t)(m0 + row) * kE + k0 + seg * 8);
            }
        } else {
            const float* A = (const float*)Ap;
#pragma unroll
            for (int j = 0; j < 4; ++j) {
                int i = tid + 256 * j;              // 128 rows x 8 x float4
                int row = i >> 3, seg = i & 7;
                aF[j] = *(const f32x4*)(A + (size_t)(m0 + row) * kE + k0 + seg * 4);
            }
        }
#pragma unroll
        for (int j = 0; j < 4; ++j) {
            int i = tid + 256 * j;                  // 128 rows x 8 x float4
            int row = i >> 3, seg = i & 7;
            wF[j] = *(const f32x4*)(W + (size_t)(n0 + row) * kE + k0 + seg * 4);
        }
    };

    auto store_tile = [&](int buf) {
        __bf16* As = AsBase + buf * 4096;
        __bf16* Ws = WsBase + buf * 4096;
        if constexpr (A_BF16) {
#pragma unroll
            for (int j = 0; j < 2; ++j) {
                int i = tid + 256 * j;
                int row = i >> 2, seg = i & 3;
                *(u32x4*)(As + row * 32 + seg * 8) = aU[j];
            }
        } else {
#pragma unroll
            for (int j = 0; j < 4; ++j) {
                int i = tid + 256 * j;
                int row = i >> 3, seg = i & 7;
                __bf16* d = As + row * 32 + seg * 4;
                d[0] = (__bf16)aF[j][0]; d[1] = (__bf16)aF[j][1];
                d[2] = (__bf16)aF[j][2]; d[3] = (__bf16)aF[j][3];
            }
        }
#pragma unroll
        for (int j = 0; j < 4; ++j) {
            int i = tid + 256 * j;
            int row = i >> 3, seg = i & 7;
            __bf16* d = Ws + row * 32 + seg * 4;
            d[0] = (__bf16)wF[j][0]; d[1] = (__bf16)wF[j][1];
            d[2] = (__bf16)wF[j][2]; d[3] = (__bf16)wF[j][3];
        }
    };

    v8f acc[2][4];
#pragma unroll
    for (int i = 0; i < 2; ++i)
#pragma unroll
        for (int j = 0; j < 4; ++j) acc[i][j] = vzero8();

    // pipeline prologue: tile 0 into buffer 0
    load_tile(0);
    store_tile(0);
    __syncthreads();

    int buf = 0;
    constexpr int KB = kE / 32;
    for (int kb = 0; kb < KB; ++kb) {
        if (kb + 1 < KB) load_tile(kb + 1);   // global loads in flight

        const __bf16* As = AsBase + buf * 4096;
        const __bf16* Ws = WsBase + buf * 4096;
        Frag af[2], bfr[4];
#pragma unroll
        for (int mt = 0; mt < 2; ++mt) {
            const __bf16* p = As + (mw0 + mt * 16 + ln) * 32 + hi * 8;
            af[mt].u[0] = *(const u32x4*)p;
            af[mt].u[1] = *(const u32x4*)(p + 16);
        }
#pragma unroll
        for (int nt = 0; nt < 4; ++nt) {
            const __bf16* p = Ws + (nw0 + nt * 16 + ln) * 32 + hi * 8;
            bfr[nt].u[0] = *(const u32x4*)p;
            bfr[nt].u[1] = *(const u32x4*)(p + 16);
        }
#pragma unroll
        for (int mt = 0; mt < 2; ++mt)
#pragma unroll
            for (int nt = 0; nt < 4; ++nt)
                acc[mt][nt] = wmma_bf16(af[mt], bfr[nt], acc[mt][nt]);

        if (kb + 1 < KB) store_tile(buf ^ 1); // waits for loads, fills other buf
        __syncthreads();
        buf ^= 1;
    }

    // ---- epilogue: bias + store ----
    float bn[4];
#pragma unroll
    for (int nt = 0; nt < 4; ++nt) bn[nt] = bias[n0 + nw0 + nt * 16 + ln];
#pragma unroll
    for (int mt = 0; mt < 2; ++mt)
#pragma unroll
        for (int nt = 0; nt < 4; ++nt)
#pragma unroll
            for (int e = 0; e < 8; ++e) {
                const int m = m0 + mw0 + mt * 16 + hi * 8 + e;
                const int n = n0 + nw0 + nt * 16 + ln;
                const float val = acc[mt][nt][e] + bn[nt];
                if constexpr (OUT_MODE == 2) {
                    ((float*)Outp)[(size_t)m * kE + n] = val;
                } else {
                    const int b = m >> 11, s = m & (kS - 1);
                    const int h = n >> 6, d = n & 63;
                    __bf16* Out = (__bf16*)Outp;
                    if constexpr (OUT_MODE == 0)
                        Out[(((size_t)(b * kH + h)) * kS + s) * kDH + d] = (__bf16)val;
                    else
                        Out[(((size_t)(b * kH + h)) * kDH + d) * kS + s] = (__bf16)val;
                }
            }
}

// ---------------------------------------------------------------------------
// Fused attention. Grid = B*(S/32). Block = 512 (16 waves, wave = head).
// LDS score buffer: Ssc[h] at h*kHS floats, row pitch 65.
//   kHS = 32*65 + 20 = 2100 so that in the mix phase both the B-fragment
//   gather (head delta 2 between lane halves -> bank offset 40) and the D
//   write-back (head delta 8 -> bank offset 32) are conflict-free, while the
//   softmax column reads (pitch 65) stay conflict-free too.
// Ps bf16 [16][32][72] (pitch 72 -> 16B-aligned b128 fragment loads).
// ---------------------------------------------------------------------------
constexpr int kHS       = 32 * 65 + 20;       // 2100 floats per head
constexpr int SSC_BYTES = kH * kHS * 4;       // 134400
constexpr int PS_BYTES  = kH * 32 * 72 * 2;   //  73728
constexpr int ATTN_SH   = SSC_BYTES + PS_BYTES + 2 * kH * 32 * 4; // 212224

__global__ __launch_bounds__(512) void attn_fused(
    const __bf16* __restrict__ Q, const __bf16* __restrict__ K,
    const __bf16* __restrict__ Vt, const float* __restrict__ Wc,
    const float* __restrict__ bc, __bf16* __restrict__ Out)
{
    extern __shared__ char smem[];
    float*  Ssc = (float*)smem;
    __bf16* Ps  = (__bf16*)(smem + SSC_BYTES);
    float*  Al  = (float*)(smem + SSC_BYTES + PS_BYTES);
    float*  Ls  = Al + kH * 32;

    const int g    = threadIdx.x >> 5;    // head id == wave id
    const int lane = threadIdx.x & 31;
    const int ln   = lane & 15;
    const int hi   = lane >> 4;

    const int b  = blockIdx.x / (kS / 32);
    const int q0 = (blockIdx.x % (kS / 32)) * 32;

    const __bf16* Qh = Q  + ((size_t)(b * kH + g) * kS + q0) * kDH;
    const __bf16* Kh = K  + (size_t)(b * kH + g) * kS * kDH;
    const __bf16* Vh = Vt + (size_t)(b * kH + g) * kDH * kS;

    // Loop-invariant Q fragments (A-matrix layout): rows contiguous in d.
    Frag qf[2][2];
#pragma unroll
    for (int mt = 0; mt < 2; ++mt)
#pragma unroll
        for (int kd = 0; kd < 2; ++kd) {
            const __bf16* p = Qh + (size_t)(mt * 16 + ln) * kDH + kd * 32 + hi * 8;
            qf[mt][kd].u[0] = *(const u32x4*)p;
            qf[mt][kd].u[1] = *(const u32x4*)(p + 16);
        }

    // Loop-invariant mixing matrix A' = I + Wc as 4 f32 WMMA A-fragments
    // (16x4 layout: element v -> K = ks*4 + hi*2 + v, row M = ln).
    v2f amix[4];
#pragma unroll
    for (int ks = 0; ks < 4; ++ks)
#pragma unroll
        for (int v = 0; v < 2; ++v) {
            const int k = ks * 4 + hi * 2 + v;
            amix[ks][v] = Wc[ln * kH + k] + ((ln == k) ? 1.0f : 0.0f);
        }
    // bc value for each D element (row = e + hi*8) of the mix result.
    float bce[8];
#pragma unroll
    for (int e = 0; e < 8; ++e) bce[e] = bc[e + hi * 8];

    float m_run = -3.0e38f, l_run = 0.0f;
    v8f o[2][4];
#pragma unroll
    for (int mt = 0; mt < 2; ++mt)
#pragma unroll
        for (int nt = 0; nt < 4; ++nt) o[mt][nt] = vzero8();

    for (int kt = 0; kt < kS / 64; ++kt) {
        const int k0 = kt * 64;

        // ---- S_raw = Q * K^T (32q x 64k): per k-slice, load all 4 K frags
        //      as one clause, then issue its 8 WMMAs. ----
        v8f sacc[2][4];
#pragma unroll
        for (int mt = 0; mt < 2; ++mt)
#pragma unroll
            for (int nt = 0; nt < 4; ++nt) sacc[mt][nt] = vzero8();
#pragma unroll
        for (int kd = 0; kd < 2; ++kd) {
            Frag kf[4];   // B-matrix: col = k-position, elems contiguous in d
#pragma unroll
            for (int nt = 0; nt < 4; ++nt) {
                const __bf16* p = Kh + (size_t)(k0 + nt * 16 + ln) * kDH + kd * 32 + hi * 8;
                kf[nt].u[0] = *(const u32x4*)p;
                kf[nt].u[1] = *(const u32x4*)(p + 16);
            }
#pragma unroll
            for (int nt = 0; nt < 4; ++nt)
#pragma unroll
                for (int mt = 0; mt < 2; ++mt)
                    sacc[mt][nt] = wmma_bf16(qf[mt][kd], kf[nt], sacc[mt][nt]);
        }

        // ---- publish scaled scores (1/sqrt(64)) to LDS ----
#pragma unroll
        for (int mt = 0; mt < 2; ++mt)
#pragma unroll
            for (int nt = 0; nt < 4; ++nt)
#pragma unroll
                for (int e = 0; e < 8; ++e) {
                    const int q = mt * 16 + hi * 8 + e;
                    Ssc[g * kHS + q * 65 + nt * 16 + ln] = sacc[mt][nt][e] * 0.125f;
                }

        // ---- issue V fragment loads NOW: they stay in flight across the
        //      whole mixing/softmax phase. ----
        Frag vf[2][4];
#pragma unroll
        for (int ks = 0; ks < 2; ++ks)
#pragma unroll
            for (int nt = 0; nt < 4; ++nt) {
                const __bf16* vp = Vh + (size_t)(nt * 16 + ln) * kS + k0 + ks * 32 + hi * 8;
                vf[ks][nt].u[0] = *(const u32x4*)vp;
                vf[ks][nt].u[1] = *(const u32x4*)(vp + 16);
            }
        if (kt + 1 < kS / 64) {  // gfx1250 global_prefetch_b8 for next tiles
            __builtin_prefetch(Kh + (size_t)(k0 + 64 + lane) * kDH, 0, 1);
            __builtin_prefetch(Kh + (size_t)(k0 + 96 + lane) * kDH, 0, 1);
            __builtin_prefetch(Vh + (size_t)lane * kS + k0 + 64, 0, 1);
            __builtin_prefetch(Vh + (size_t)(lane + 32) * kS + k0 + 64, 0, 1);
        }
        __syncthreads();

        // ---- cross-head mixing on the f32 WMMA unit, in place. ----
        // mixed[g'][pos] = sum_h (I+Wc)[g'][h] * scores[h][pos] + bc[g'].
        // Wave g owns query rows {2g, 2g+1}; 4 chunks of 16 keys per row.
        // In-place is safe: each wave reads and writes only its own slice,
        // and a chunk's gathers complete before its write-back (data dep).
#pragma unroll
        for (int r = 0; r < 2; ++r) {
            const int q = 2 * g + r;
#pragma unroll
            for (int c = 0; c < 4; ++c) {
                const int pos = q * 65 + c * 16 + ln;
                v8f mix = vzero8();
#pragma unroll
                for (int ks = 0; ks < 4; ++ks) {
                    v2f bfrag;   // B 4x16: element v -> K = ks*4 + hi*2 + v
#pragma unroll
                    for (int v = 0; v < 2; ++v)
                        bfrag[v] = Ssc[(ks * 4 + hi * 2 + v) * kHS + pos];
                    mix = wmma_f32k4(amix[ks], bfrag, mix);
                }
#pragma unroll
                for (int e = 0; e < 8; ++e)
                    Ssc[(e + hi * 8) * kHS + pos] = mix[e] + bce[e];
            }
        }
        __syncthreads();

        // ---- online softmax (lane = query row); scores already mixed ----
        float tmax = -3.0e38f;
#pragma unroll
        for (int kk = 0; kk < 64; ++kk)
            tmax = fmaxf(tmax, Ssc[g * kHS + lane * 65 + kk]);
        const float m_new = fmaxf(m_run, tmax);
        const float alpha = __expf(m_run - m_new);
        float psum = 0.0f;
#pragma unroll
        for (int kk = 0; kk < 64; ++kk) {
            const float pv = __expf(Ssc[g * kHS + lane * 65 + kk] - m_new);
            psum += pv;
            Ps[(size_t)(g * 32 + lane) * 72 + kk] = (__bf16)pv;
        }
        l_run = l_run * alpha + psum;
        m_run = m_new;
        Al[g * 32 + lane] = alpha;
        __syncthreads();

        // ---- rescale accumulator, then O += P * V (16 WMMAs, V preloaded) --
#pragma unroll
        for (int mt = 0; mt < 2; ++mt) {
            float av[8];
#pragma unroll
            for (int e = 0; e < 8; ++e) av[e] = Al[g * 32 + mt * 16 + hi * 8 + e];
#pragma unroll
            for (int nt = 0; nt < 4; ++nt)
#pragma unroll
                for (int e = 0; e < 8; ++e) o[mt][nt][e] *= av[e];
        }
#pragma unroll
        for (int ks = 0; ks < 2; ++ks) {
            Frag pf[2];
#pragma unroll
            for (int mt = 0; mt < 2; ++mt) {
                const __bf16* p = Ps + (size_t)(g * 32 + mt * 16 + ln) * 72 + ks * 32 + hi * 8;
                pf[mt].u[0] = *(const u32x4*)p;
                pf[mt].u[1] = *(const u32x4*)(p + 16);
            }
#pragma unroll
            for (int nt = 0; nt < 4; ++nt)
#pragma unroll
                for (int mt = 0; mt < 2; ++mt)
                    o[mt][nt] = wmma_bf16(pf[mt], vf[ks][nt], o[mt][nt]);
        }
    }

    // ---- final normalization and store: out[b][q][g*64+d] (bf16) ----
    Ls[g * 32 + lane] = l_run;
    __syncthreads();
    __bf16* Oh = Out + ((size_t)b * kS + q0) * kE + g * kDH;
#pragma unroll
    for (int mt = 0; mt < 2; ++mt) {
        float lv[8];
#pragma unroll
        for (int e = 0; e < 8; ++e) lv[e] = Ls[g * 32 + mt * 16 + hi * 8 + e];
#pragma unroll
        for (int nt = 0; nt < 4; ++nt)
#pragma unroll
            for (int e = 0; e < 8; ++e)
                Oh[(size_t)(mt * 16 + hi * 8 + e) * kE + nt * 16 + ln] =
                    (__bf16)(o[mt][nt][e] / lv[e]);
    }
}

// ---------------------------------------------------------------------------
extern "C" void kernel_launch(void* const* d_in, const int* in_sizes, int n_in,
                              void* d_out, int out_size, void* d_ws, size_t ws_size,
                              hipStream_t stream)
{
    (void)in_sizes; (void)n_in; (void)out_size; (void)ws_size;
    const float* query = (const float*)d_in[0];
    const float* key_  = (const float*)d_in[1];
    const float* value = (const float*)d_in[2];
    const float* Wq = (const float*)d_in[3];
    const float* bq = (const float*)d_in[4];
    const float* Wk = (const float*)d_in[5];
    const float* bk = (const float*)d_in[6];
    const float* Wv = (const float*)d_in[7];
    const float* bv = (const float*)d_in[8];
    const float* Wc = (const float*)d_in[9];
    const float* bc = (const float*)d_in[10];
    const float* Wo = (const float*)d_in[11];
    const float* bo = (const float*)d_in[12];

    // Workspace: 4 x 8 MB bf16 tensors = 32 MB.
    const size_t NE = (size_t)kB * kH * kS * kDH;  // 4 Mi elements
    __bf16* Qbf = (__bf16*)d_ws;
    __bf16* Kbf = Qbf + NE;
    __bf16* Vt  = Kbf + NE;   // [B,H,DH,S]
    __bf16* AO  = Vt  + NE;   // attention output, [B*S, E] bf16

    constexpr int GEMM_SH = 2 * (128 * 32 + 128 * 32) * 2;  // 32 KB
    // hipFuncSetAttribute is not stream-ordered; safe under graph capture.
    hipFuncSetAttribute((const void*)attn_fused,
                        hipFuncAttributeMaxDynamicSharedMemorySize, ATTN_SH);

    dim3 ggrid(256), gblk(256);
    gemm_bf16_wmma<0, false><<<ggrid, gblk, GEMM_SH, stream>>>(query, Wq, bq, Qbf);
    gemm_bf16_wmma<0, false><<<ggrid, gblk, GEMM_SH, stream>>>(key_,  Wk, bk, Kbf);
    gemm_bf16_wmma<1, false><<<ggrid, gblk, GEMM_SH, stream>>>(value, Wv, bv, Vt);

    attn_fused<<<dim3(kB * (kS / 32)), dim3(512), ATTN_SH, stream>>>(
        Qbf, Kbf, Vt, Wc, bc, AO);

    gemm_bf16_wmma<2, true><<<ggrid, gblk, GEMM_SH, stream>>>(AO, Wo, bo, d_out);
}
